// GCNReg_0mlp_29703993819337
// MI455X (gfx1250) — matrix operand order repaired and checked by reading, not compiled, not run on hardware
//
#include <hip/hip_runtime.h>
#include <hip/hip_bf16.h>

typedef __attribute__((ext_vector_type(16))) __bf16 v16bf;
typedef __attribute__((ext_vector_type(8)))  __bf16 v8bf;
typedef __attribute__((ext_vector_type(8)))  float  v8f;

#define HDIM 128
#define LDSPAD 8                 // 8 bf16 = 16B pad: keeps 16B alignment, shifts banks by 4/row
#define LDW (HDIM + LDSPAD)
#define ROWS_PB 128              // rows of X per block
#define G_GRAPHS 512

union BF16frag { v16bf v; v8bf h[2]; __bf16 e[16]; };
union F32x8    { v8f   v; float e[8]; };

// ---------------- degree / norm ----------------
__global__ void deg_kernel(const int* __restrict__ src, const int* __restrict__ dst,
                           float* __restrict__ deg_out, float* __restrict__ deg_in, int E) {
    int e = blockIdx.x * blockDim.x + threadIdx.x;
    if (e < E) {
        atomicAdd(&deg_out[src[e]], 1.0f);
        atomicAdd(&deg_in[dst[e]], 1.0f);
    }
}

__global__ void norm_kernel(float* __restrict__ deg, int n) {
    int i = blockIdx.x * blockDim.x + threadIdx.x;
    if (i < n) deg[i] = rsqrtf(fmaxf(deg[i], 1.0f));
}

// ---------------- WMMA GEMM:  Y = (X * norm_row) @ W   (N x 128) @ (128 x 128) ----------------
// Block = 256 threads = 8 waves; block covers ROWS_PB rows.  Wave w owns cols [16w,16w+16).
// Each wave processes TWO row-tiles concurrently (independent accumulators sharing the
// B fragment) so consecutive WMMAs never chain on the same D registers.
__global__ __launch_bounds__(256)
void gemm_norm_bf16_wmma(const float* __restrict__ X, const float* __restrict__ norm,
                         const float* __restrict__ W, float* __restrict__ Y, int N) {
    __shared__ __bf16 sA [ROWS_PB][LDW];   // norm-scaled A tile        (~34.8 KB)
    __shared__ __bf16 sWt[HDIM][LDW];      // W transposed: sWt[col][k] (~34.8 KB)

    const int tid = threadIdx.x;
    const int m0  = blockIdx.x * ROWS_PB;
    const bool full_tile = (m0 + ROWS_PB) <= N;

    // stage W transposed -> bf16 LDS (W row-major [k][col])
    for (int i = tid; i < HDIM * HDIM; i += 256) {
        int k = i / HDIM, c = i % HDIM;
        sWt[c][k] = (__bf16)W[i];
    }
    // stage A tile, scaled by per-row norm (contiguous global reads)
    for (int i = tid; i < ROWS_PB * HDIM; i += 256) {
        int r = i / HDIM, k = i % HDIM;
        int row = m0 + r;
        float v = (row < N) ? X[(long long)row * HDIM + k] * norm[row] : 0.0f;
        sA[r][k] = (__bf16)v;
    }
    __syncthreads();

    const int wave = tid >> 5;
    const int lane = tid & 31;
    const int n0   = wave * 16;
    const int hi   = lane >> 4;          // lane half: ISA 7.12.2 K-split
    const int rl   = lane & 15;

    for (int mt = 0; mt < ROWS_PB; mt += 32) {
        v8f acc0 = {}, acc1 = {};
        #pragma unroll
        for (int kt = 0; kt < HDIM; kt += 32) {
            BF16frag a0, a1, b;
            // A 16x32: lanes 0-15 K {0..7,16..23}; lanes 16-31 K {8..15,24..31}; two 16B DS loads
            const __bf16* pa0 = &sA[mt +      rl][kt + (hi ? 8 : 0)];
            const __bf16* pa1 = &sA[mt + 16 + rl][kt + (hi ? 8 : 0)];
            a0.h[0] = *(const v8bf*)(pa0);
            a0.h[1] = *(const v8bf*)(pa0 + 16);
            a1.h[0] = *(const v8bf*)(pa1);
            a1.h[1] = *(const v8bf*)(pa1 + 16);
            // B 32x16 (shared by both row tiles): lanes 0-15 K {0..7,16..23}; lanes 16-31 K {8..15,24..31}
            const __bf16* pb = &sWt[n0 + rl][kt + (hi ? 8 : 0)];
            b.h[0] = *(const v8bf*)(pb);
            b.h[1] = *(const v8bf*)(pb + 16);

            acc0 = __builtin_amdgcn_wmma_f32_16x16x32_bf16(
                       false, a0.v, false, b.v, (short)0, acc0, false, false);
            acc1 = __builtin_amdgcn_wmma_f32_16x16x32_bf16(
                       false, a1.v, false, b.v, (short)0, acc1, false, false);
        }
        // C/D layout: VGPR r -> row (tile_base + r + 8*hi), col = n0 + (lane&15)
        F32x8 r0; r0.v = acc0;
        F32x8 r1; r1.v = acc1;
        const int row0 = m0 + mt + (hi ? 8 : 0);
        float* py0 = Y + ((long long)row0 * HDIM + n0 + rl);
        float* py1 = py0 + 16 * HDIM;
        if (full_tile) {
            #pragma unroll
            for (int r = 0; r < 8; ++r) py0[r * HDIM] = r0.e[r];
            #pragma unroll
            for (int r = 0; r < 8; ++r) py1[r * HDIM] = r1.e[r];
        } else {
            #pragma unroll
            for (int r = 0; r < 8; ++r)
                if (row0 + r < N) py0[r * HDIM] = r0.e[r];
            #pragma unroll
            for (int r = 0; r < 8; ++r)
                if (row0 + 16 + r < N) py1[r * HDIM] = r1.e[r];
        }
    }
}

// ---------------- edge scatter:  AGG[dst] += XP[src]  ----------------
__global__ void scatter_kernel(const float* __restrict__ XP, const int* __restrict__ src,
                               const int* __restrict__ dst, float* __restrict__ AGG, int E) {
    long long tid = (long long)blockIdx.x * blockDim.x + threadIdx.x;
    long long total = (long long)E * (HDIM / 4);
    if (tid >= total) return;
    int e = (int)(tid >> 5);            // HDIM/4 == 32 chunks per edge
    int f = ((int)tid & 31) * 4;
    const float4 v = *(const float4*)(XP + (long long)src[e] * HDIM + f);
    float* o = AGG + (long long)dst[e] * HDIM + f;
    atomicAdd(o + 0, v.x);
    atomicAdd(o + 1, v.y);
    atomicAdd(o + 2, v.z);
    atomicAdd(o + 3, v.w);
}

// ---------------- epilogue: OUT = relu(AGG * norm_dst + b) ----------------
__global__ void post_relu_kernel(const float* __restrict__ AGG, const float* __restrict__ norm_dst,
                                 const float* __restrict__ b, float* __restrict__ OUT, int N) {
    long long i = (long long)blockIdx.x * blockDim.x + threadIdx.x;
    if (i >= (long long)N * HDIM) return;
    int row = (int)(i / HDIM);
    int col = (int)(i % HDIM);
    float v = AGG[i] * norm_dst[row] + b[col];
    OUT[i] = fmaxf(v, 0.0f);
}

// ---------------- per-graph mean pooling (sums + counts via atomics) ----------------
__global__ void pool_kernel(const float* __restrict__ h2, const int* __restrict__ gid,
                            float* __restrict__ gsums, float* __restrict__ gcnt, int N) {
    long long tid = (long long)blockIdx.x * blockDim.x + threadIdx.x;
    if (tid >= (long long)N * (HDIM / 4)) return;
    int node = (int)(tid >> 5);
    int f = ((int)tid & 31) * 4;
    int g = gid[node];
    const float4 v = *(const float4*)(h2 + (long long)node * HDIM + f);
    float* o = gsums + (long long)g * HDIM + f;
    atomicAdd(o + 0, v.x);
    atomicAdd(o + 1, v.y);
    atomicAdd(o + 2, v.z);
    atomicAdd(o + 3, v.w);
    if (f == 0) atomicAdd(&gcnt[g], 1.0f);
}

// ---------------- final: out = (gsums/cnt) @ W3 + b3 ----------------
__global__ void final_kernel(const float* __restrict__ gsums, const float* __restrict__ gcnt,
                             const float* __restrict__ W3, const float* __restrict__ b3,
                             float* __restrict__ out, int G, int C) {
    int g = blockIdx.x * blockDim.x + threadIdx.x;
    if (g >= G) return;
    float inv = 1.0f / fmaxf(gcnt[g], 1.0f);
    for (int c = 0; c < C; ++c) {
        float acc = 0.0f;
        for (int k = 0; k < HDIM; ++k)
            acc += gsums[(long long)g * HDIM + k] * inv * W3[k * C + c];
        out[g * C + c] = acc + b3[c];
    }
}

extern "C" void kernel_launch(void* const* d_in, const int* in_sizes, int n_in,
                              void* d_out, int out_size, void* d_ws, size_t ws_size,
                              hipStream_t stream) {
    const float* h   = (const float*)d_in[0];
    const int*   src = (const int*)  d_in[1];
    const int*   dst = (const int*)  d_in[2];
    const int*   gid = (const int*)  d_in[3];
    // d_in[4] = num_graphs scalar (device); G known statically
    const float* W1 = (const float*)d_in[5];
    const float* b1 = (const float*)d_in[6];
    const float* W2 = (const float*)d_in[7];
    const float* b2 = (const float*)d_in[8];
    const float* W3 = (const float*)d_in[9];
    const float* b3 = (const float*)d_in[10];

    const int N = in_sizes[0] / HDIM;
    const int E = in_sizes[1];
    const int G = G_GRAPHS;
    const int C = out_size / G;

    // workspace carve (floats)
    float* ws = (float*)d_ws;
    float* norm_src = ws;                 ws += N;            // deg_out -> norm_src (in place)
    float* norm_dst = ws;                 ws += N;            // deg_in  -> norm_dst (in place)
    float* bufA     = ws;                 ws += (size_t)N * HDIM;
    float* bufB     = ws;                 ws += (size_t)N * HDIM;
    float* gsums    = ws;                 ws += (size_t)G * HDIM;
    float* gcnt     = ws;                 ws += G;

    const int T = 256;
    const unsigned ebl  = (unsigned)((E + T - 1) / T);
    const unsigned nh   = (unsigned)(((long long)N * HDIM + T - 1) / T);
    const unsigned sct  = (unsigned)(((long long)E * (HDIM / 4) + T - 1) / T);
    const unsigned pol  = (unsigned)(((long long)N * (HDIM / 4) + T - 1) / T);
    const unsigned rows = (unsigned)((N + ROWS_PB - 1) / ROWS_PB);

    // degrees -> norms
    hipMemsetAsync(norm_src, 0, sizeof(float) * 2 * (size_t)N, stream);  // contiguous pair
    deg_kernel<<<ebl, T, 0, stream>>>(src, dst, norm_src, norm_dst, E);
    norm_kernel<<<(2 * N + T - 1) / T, T, 0, stream>>>(norm_src, 2 * N);

    // ---- layer 1 ----
    gemm_norm_bf16_wmma<<<rows, T, 0, stream>>>(h, norm_src, W1, bufA, N);
    hipMemsetAsync(bufB, 0, sizeof(float) * (size_t)N * HDIM, stream);
    scatter_kernel<<<sct, T, 0, stream>>>(bufA, src, dst, bufB, E);
    post_relu_kernel<<<nh, T, 0, stream>>>(bufB, norm_dst, b1, bufB, N);   // h1 in bufB

    // ---- layer 2 ----
    gemm_norm_bf16_wmma<<<rows, T, 0, stream>>>(bufB, norm_src, W2, bufA, N);
    hipMemsetAsync(bufB, 0, sizeof(float) * (size_t)N * HDIM, stream);     // h1 dead after gemm
    scatter_kernel<<<sct, T, 0, stream>>>(bufA, src, dst, bufB, E);
    post_relu_kernel<<<nh, T, 0, stream>>>(bufB, norm_dst, b2, bufB, N);   // h2 in bufB

    // ---- readout ----
    hipMemsetAsync(gsums, 0, sizeof(float) * ((size_t)G * HDIM + G), stream); // sums + counts contiguous
    pool_kernel<<<pol, T, 0, stream>>>(bufB, gid, gsums, gcnt, N);
    final_kernel<<<(G + T - 1) / T, T, 0, stream>>>(gsums, gcnt, W3, b3, (float*)d_out, G, C);
}
